// MultiHeadAttention_7172595384931
// MI455X (gfx1250) — matrix operand order repaired and checked by reading, not compiled, not run on hardware
//
#include <hip/hip_runtime.h>
#include <hip/hip_bf16.h>

#define D_MODEL 1024
#define N_HEADS 16
#define DEPTH   64
#define BATCH   2
#define SEQ     2048
#define M_TOT   (BATCH * SEQ)   // 4096

typedef __attribute__((ext_vector_type(16))) __bf16 v16bf;
typedef __attribute__((ext_vector_type(8)))  float  v8f;
typedef __attribute__((ext_vector_type(4)))  unsigned int u32x4;
typedef __attribute__((ext_vector_type(8)))  int          i32x8;
typedef __attribute__((ext_vector_type(4)))  int          i32x4;

#if defined(__gfx1250__) && __has_builtin(__builtin_amdgcn_tensor_load_to_lds)
#define HAVE_TDM 1
#endif

// Compile-time probes (driver returns stderr; confirms which paths are live)
#if defined(__HIP_DEVICE_COMPILE__)
#if defined(HAVE_TDM)
#pragma message("CDNA5-PROBE: attn staging = TDM tensor_load_to_lds (ENABLED)")
#else
#pragma message("CDNA5-PROBE: attn staging = cooperative-copy fallback (TDM builtin absent)")
#endif
#if __has_builtin(__builtin_amdgcn_permlane16)
#pragma message("CDNA5-PROBE: softmax reductions = v_permlane16_b32 (ENABLED)")
#else
#pragma message("CDNA5-PROBE: softmax reductions = __shfl_xor fallback")
#endif
#endif

// ---------------------------------------------------------------------------
// Fragment loaders matching CDNA5 WMMA VGPR layouts (ISA 7.12.2, wave32).
// A 16x32 (bf16): lane L holds row (L&15); halves[0..7] = K[(L>>4)*8 ..],
//                 halves[8..15] = K[(L>>4)*8 + 16 ..].
// B 32x16 (bf16): lane L holds column (L&15); same K order as A.
// ---------------------------------------------------------------------------
__device__ inline v16bf load_a_frag(const __bf16* base, int stride) {
  int lane = threadIdx.x & 31;
  const __bf16* p = base + (lane & 15) * stride + ((lane >> 4) * 8);
  v16bf a;
#pragma unroll
  for (int i = 0; i < 8; ++i) { a[i] = p[i]; a[8 + i] = p[16 + i]; }
  return a;
}

__device__ inline v16bf load_b_frag(const __bf16* base, int stride) {
  int lane = threadIdx.x & 31;
  const __bf16* p = base + (lane & 15) * stride + ((lane >> 4) * 8);
  v16bf b;
#pragma unroll
  for (int i = 0; i < 8; ++i) { b[i] = p[i]; b[8 + i] = p[16 + i]; }
  return b;
}

__device__ inline v8f wmma_bf16(v16bf a, v16bf b, v8f c) {
  return __builtin_amdgcn_wmma_f32_16x16x32_bf16(false, a, false, b,
                                                 (short)0, c, false, false);
}

// ---------------------------------------------------------------------------
// 16-lane xor-butterfly reductions. Prefer v_permlane16_b32 (pure VALU) over
// __shfl_xor (which lowers to ds_bpermute and competes with WMMA fragment
// traffic for LDS bandwidth). Selector nibbles encode lane^mask per row of 16.
// ---------------------------------------------------------------------------
#if __has_builtin(__builtin_amdgcn_permlane16)
__device__ inline float xor16(float v, unsigned lo, unsigned hi) {
  unsigned s = __float_as_uint(v);
  return __uint_as_float(__builtin_amdgcn_permlane16(s, s, lo, hi, false, false));
}
__device__ inline float redmax16(float v) {
  v = fmaxf(v, xor16(v, 0x67452301u, 0xEFCDAB89u));  // ^1
  v = fmaxf(v, xor16(v, 0x54761032u, 0xDCFE98BAu));  // ^2
  v = fmaxf(v, xor16(v, 0x32107654u, 0xBA98FEDCu));  // ^4
  v = fmaxf(v, xor16(v, 0xFEDCBA98u, 0x76543210u));  // ^8
  return v;
}
__device__ inline float redsum16(float v) {
  v += xor16(v, 0x67452301u, 0xEFCDAB89u);
  v += xor16(v, 0x54761032u, 0xDCFE98BAu);
  v += xor16(v, 0x32107654u, 0xBA98FEDCu);
  v += xor16(v, 0xFEDCBA98u, 0x76543210u);
  return v;
}
#else
__device__ inline float redmax16(float v) {
  for (int off = 8; off >= 1; off >>= 1) v = fmaxf(v, __shfl_xor(v, off, 16));
  return v;
}
__device__ inline float redsum16(float v) {
  for (int off = 8; off >= 1; off >>= 1) v += __shfl_xor(v, off, 16);
  return v;
}
#endif

// ---------------------------------------------------------------------------
// Tensor Data Mover: issue a 2-D bf16 tile load (D# per ISA 8.3/8.4).
// group0: count=1 | lds_addr | global_addr | type=2.  group1 passed in.
// ---------------------------------------------------------------------------
#if defined(HAVE_TDM)
__device__ inline void tdm_load_tile(unsigned lds_addr, const void* gaddr,
                                     i32x8 g1) {
  unsigned long long ga = (unsigned long long)gaddr;
  u32x4 g0 = { 1u,                                   // count=1, user mode
               lds_addr,                             // [63:32] LDS byte addr
               (unsigned)ga,                         // [95:64] gaddr lo
               (unsigned)(ga >> 32) | 0x80000000u }; // [120:96] hi | type=2
  i32x4 z4 = { 0, 0, 0, 0 };
#if __clang_major__ >= 23
  i32x8 z8 = { 0, 0, 0, 0, 0, 0, 0, 0 };
  __builtin_amdgcn_tensor_load_to_lds(g0, g1, z4, z4, z8, 0);
#else
  __builtin_amdgcn_tensor_load_to_lds(g0, g1, z4, z4, 0);
#endif
}
// group1 word0: data_size=1(2B) | pad_enable | pad_interval=32dw | pad_amt=4dw
// (pad reproduces the +8-half row padding -> row stride 72 halves in LDS)
#define TDM_G1_W0 ((int)((1u << 16) | (1u << 20) | (4u << 22) | (3u << 25)))
#endif

// ---------------------------------------------------------------------------
// Generic  C = A @ W^T + bias  (M=4096, N=1024, K=1024), bf16 WMMA, f32 acc.
// MODE 0: out bf16, head-scattered [B,H,S,64], value scaled by `scale` (q/k)
// MODE 1: out bf16, transposed heads [B,H,64,S]                        (v)
// MODE 2: out f32 row-major [M,N]                                      (o)
// Block: 256 thr (8 waves), tile 128x128, wave tile 32x64, K-step 32.
// ---------------------------------------------------------------------------
template <typename TA, int MODE>
__global__ __launch_bounds__(256) void gemm_xwT(const TA* __restrict__ A,
                                                const float* __restrict__ W,
                                                const float* __restrict__ bias,
                                                void* __restrict__ Out,
                                                float scale) {
  constexpr int BM = 128, BN = 128, BK = 32;
  constexpr int LDA = BK + 8, LDB = BK + 8;  // +8 halves: bank-conflict pad
  __shared__ __align__(16) __bf16 sA[BM * LDA];
  __shared__ __align__(16) __bf16 sB[BN * LDB];

  const int tid = threadIdx.x, lane = tid & 31, wid = tid >> 5;
  const int wm = wid & 3, wn = wid >> 2;    // 4x2 wave grid
  const int m0 = blockIdx.x * BM;
  const int n0 = blockIdx.y * BN;

  v8f acc[2][4] = {};
  float bj[4];
#pragma unroll
  for (int j = 0; j < 4; ++j)
    bj[j] = bias[n0 + wn * 64 + j * 16 + (lane & 15)];

  for (int k0 = 0; k0 < D_MODEL; k0 += BK) {
    __syncthreads();
    {
      const int tc = tid & 7;   // 8 col-groups of 4 elements
      const int tr = tid >> 3;  // 32 rows per pass
#pragma unroll
      for (int r = 0; r < 4; ++r) {  // A tile: 128x32
        const int row = tr + r * 32;
        const TA* src = A + (size_t)(m0 + row) * D_MODEL + k0 + tc * 4;
        __bf16* dst = &sA[row * LDA + tc * 4];
#pragma unroll
        for (int e = 0; e < 4; ++e) dst[e] = (__bf16)(float)src[e];
      }
#pragma unroll
      for (int r = 0; r < 4; ++r) {  // W tile: 128x32
        const int row = tr + r * 32;
        const float* src = W + (size_t)(n0 + row) * D_MODEL + k0 + tc * 4;
        __bf16* dst = &sB[row * LDB + tc * 4];
#pragma unroll
        for (int e = 0; e < 4; ++e) dst[e] = (__bf16)src[e];
      }
    }
    __syncthreads();

    v16bf af[2], bf[4];
#pragma unroll
    for (int i = 0; i < 2; ++i)
      af[i] = load_a_frag(&sA[(wm * 32 + i * 16) * LDA], LDA);
#pragma unroll
    for (int j = 0; j < 4; ++j)
      bf[j] = load_b_frag(&sB[(wn * 64 + j * 16) * LDB], LDB);
#pragma unroll
    for (int i = 0; i < 2; ++i)
#pragma unroll
      for (int j = 0; j < 4; ++j)
        acc[i][j] = wmma_bf16(af[i], bf[j], acc[i][j]);
  }

  // Epilogue: C layout -> VGPR r holds rows (r, r+8) across the lane halves.
  const int lr = lane >> 4, lc = lane & 15;
#pragma unroll
  for (int i = 0; i < 2; ++i)
#pragma unroll
    for (int j = 0; j < 4; ++j)
#pragma unroll
      for (int r = 0; r < 8; ++r) {
        const int m = m0 + wm * 32 + i * 16 + r + lr * 8;
        const int n = n0 + wn * 64 + j * 16 + lc;
        float v = acc[i][j][r] + bj[j];
        if (MODE == 2) {
          ((float*)Out)[(size_t)m * D_MODEL + n] = v;
        } else {
          v *= scale;
          const int b = m >> 11, s = m & (SEQ - 1);
          const int h = n >> 6, d = n & (DEPTH - 1);
          __bf16* O = (__bf16*)Out;
          size_t idx;
          if (MODE == 0)
            idx = (((size_t)(b * N_HEADS + h)) * SEQ + s) * DEPTH + d;
          else
            idx = (((size_t)(b * N_HEADS + h)) * DEPTH + d) * SEQ + s;
          O[idx] = (__bf16)v;
        }
      }
}

// ---------------------------------------------------------------------------
// Flash-style attention: block = 4 waves, 64 query rows; each wave owns 16
// rows (q resident in VGPRs). Per 64-key chunk: TDM (or fallback copy) stages
// K [64x64] and V^T [64x64] bf16 tiles in LDS, 8 score WMMAs, online softmax
// (permlane16 row reductions), p -> LDS -> A-frags, 8 output WMMAs.
// q already carries the 1/sqrt(depth) scale.
// ---------------------------------------------------------------------------
__global__ __launch_bounds__(128) void attn_kernel(const __bf16* __restrict__ Qh,
                                                   const __bf16* __restrict__ Kh,
                                                   const __bf16* __restrict__ Vt,
                                                   __bf16* __restrict__ Oc) {
  constexpr int KT = 64;
  constexpr int LK = DEPTH + 8;  // kbuf row stride (halves) = 72
  constexpr int LV = KT + 8;     // vbuf row stride = 72
  constexpr int LP = KT + 8;     // pbuf row stride = 72
  __shared__ __align__(16) __bf16 kbuf[KT * LK];       // [key][depth]
  __shared__ __align__(16) __bf16 vbuf[DEPTH * LV];    // [depth][key]
  __shared__ __align__(16) __bf16 pbuf[4 * 16 * LP];   // per-wave p tiles

  const int tid = threadIdx.x, lane = tid & 31, wid = tid >> 5;
  const int bh = blockIdx.y;           // b*16 + h
  const int m0 = blockIdx.x * 64;
  const __bf16* qp = Qh + (size_t)bh * SEQ * DEPTH;
  const __bf16* kp = Kh + (size_t)bh * SEQ * DEPTH;
  const __bf16* vp = Vt + (size_t)bh * DEPTH * SEQ;

  // q fragments: 16 rows x depth 64 = two 16x32 A-frags, resident in VGPRs.
  const int qrow0 = m0 + wid * 16;
  v16bf aq[2];
  {
    const __bf16* p = qp + (size_t)(qrow0 + (lane & 15)) * DEPTH;
    const int koff = (lane >> 4) * 8;
#pragma unroll
    for (int kk = 0; kk < 2; ++kk)
#pragma unroll
      for (int i = 0; i < 8; ++i) {
        aq[kk][i]     = p[kk * 32 + koff + i];
        aq[kk][8 + i] = p[kk * 32 + koff + 16 + i];
      }
  }

#if defined(HAVE_TDM)
  // D# group1 for K tile: tensor [dim0=64, dim1=2048] stride0=64,
  // tile 64x64, stride1 = 64*2048 = 131072 (words 6/7).
  const i32x8 g1k = { TDM_G1_W0, 64 << 16, (int)(2048u << 16), 64 << 16,
                      64, 64, 0, 2 };
  // D# group1 for V^T tile: tensor [dim0=2048, dim1=64] stride0=2048,
  // tile 64x64, stride1 = 131072.
  const i32x8 g1v = { TDM_G1_W0, (int)(2048u << 16), 64 << 16, 64 << 16,
                      64, 2048, 0, 2 };
  const unsigned kbo = (unsigned)(unsigned long long)(void*)kbuf;
  const unsigned vbo = (unsigned)(unsigned long long)(void*)vbuf;
#endif

  float mrun[8], lrun[8];
#pragma unroll
  for (int r = 0; r < 8; ++r) { mrun[r] = -3.0e38f; lrun[r] = 0.f; }
  v8f o[4] = {};

  for (int s0 = 0; s0 < SEQ; s0 += KT) {
    __syncthreads();
#if defined(HAVE_TDM)
    if (wid == 0) {  // one wave drives the Tensor Data Mover
      tdm_load_tile(kbo, kp + (size_t)s0 * DEPTH, g1k);
      tdm_load_tile(vbo, vp + s0, g1v);
      __builtin_amdgcn_s_wait_tensorcnt(0);
    }
#else
    {  // fallback: cooperative copy, 64 rows x 64 halves each tile
      const int kr = tid >> 1, c = (tid & 1) * 32;
      const __bf16* src = kp + (size_t)(s0 + kr) * DEPTH + c;
      __bf16* dst = &kbuf[kr * LK + c];
#pragma unroll
      for (int i = 0; i < 32; ++i) dst[i] = src[i];
      const __bf16* src2 = vp + (size_t)kr * SEQ + s0 + c;
      __bf16* dst2 = &vbuf[kr * LV + c];
#pragma unroll
      for (int i = 0; i < 32; ++i) dst2[i] = src2[i];
    }
#endif
    __syncthreads();

    // scores: 16 rows x 64 keys, K=depth 64 in two steps -> 8 WMMAs
    v8f sc[4] = {};
#pragma unroll
    for (int kk = 0; kk < 2; ++kk)
#pragma unroll
      for (int j = 0; j < 4; ++j) {
        v16bf bfrag = load_b_frag(&kbuf[(j * 16) * LK + kk * 32], LK);
        sc[j] = wmma_bf16(aq[kk], bfrag, sc[j]);
      }

    // online softmax; a row spans the 16 lanes of a half-wave
    float pscale[8];
#pragma unroll
    for (int r = 0; r < 8; ++r) {
      float t = fmaxf(fmaxf(sc[0][r], sc[1][r]), fmaxf(sc[2][r], sc[3][r]));
      t = redmax16(t);
      const float mnew = fmaxf(mrun[r], t);
      pscale[r] = __expf(mrun[r] - mnew);
      mrun[r] = mnew;
    }
#pragma unroll
    for (int j = 0; j < 4; ++j)
#pragma unroll
      for (int r = 0; r < 8; ++r) sc[j][r] = __expf(sc[j][r] - mrun[r]);
#pragma unroll
    for (int r = 0; r < 8; ++r) {
      float t = (sc[0][r] + sc[1][r]) + (sc[2][r] + sc[3][r]);
      lrun[r] = lrun[r] * pscale[r] + redsum16(t);
    }
#pragma unroll
    for (int dd = 0; dd < 4; ++dd)
#pragma unroll
      for (int r = 0; r < 8; ++r) o[dd][r] *= pscale[r];

    // p: C-layout -> LDS -> A-fragment layout (per-wave region; same-wave DS
    // ops are in-order, no block barrier needed)
    {
      __bf16* pb = &pbuf[wid * 16 * LP];
      const int lr = lane >> 4, lc = lane & 15;
#pragma unroll
      for (int j = 0; j < 4; ++j)
#pragma unroll
        for (int r = 0; r < 8; ++r)
          pb[(r + lr * 8) * LP + j * 16 + lc] = (__bf16)sc[j][r];
    }
    v16bf pfrag[2];
#pragma unroll
    for (int kk = 0; kk < 2; ++kk)
      pfrag[kk] = load_a_frag(&pbuf[wid * 16 * LP + kk * 32], LP);
#pragma unroll
    for (int dd = 0; dd < 4; ++dd)
#pragma unroll
      for (int kk = 0; kk < 2; ++kk) {
        v16bf bv = load_b_frag(&vbuf[(dd * 16) * LV + kk * 32], LV);
        o[dd] = wmma_bf16(pfrag[kk], bv, o[dd]);
      }
  }

  // epilogue: normalize and write combined-head layout [B,S,D_MODEL] (bf16)
  const int b = bh >> 4, h = bh & 15;
  const int lr = lane >> 4, lc = lane & 15;
#pragma unroll
  for (int dd = 0; dd < 4; ++dd)
#pragma unroll
    for (int r = 0; r < 8; ++r) {
      const int srow = qrow0 + r + lr * 8;
      const int col = h * DEPTH + dd * 16 + lc;
      const float v = o[dd][r] / lrun[r];
      Oc[((size_t)b * SEQ + srow) * D_MODEL + col] = (__bf16)v;
    }
}

// ---------------------------------------------------------------------------
extern "C" void kernel_launch(void* const* d_in, const int* in_sizes, int n_in,
                              void* d_out, int out_size, void* d_ws,
                              size_t ws_size, hipStream_t stream) {
  const float* Q  = (const float*)d_in[0];
  const float* K  = (const float*)d_in[1];
  const float* V  = (const float*)d_in[2];
  const float* Wq = (const float*)d_in[3];
  const float* bq = (const float*)d_in[4];
  const float* Wk = (const float*)d_in[5];
  const float* bk = (const float*)d_in[6];
  const float* Wv = (const float*)d_in[7];
  const float* bv = (const float*)d_in[8];
  const float* Wo = (const float*)d_in[9];
  const float* bo = (const float*)d_in[10];
  float* out = (float*)d_out;

  char* ws = (char*)d_ws;
  const size_t SZ = (size_t)BATCH * N_HEADS * SEQ * DEPTH * 2;  // 8 MB each
  __bf16* qb  = (__bf16*)(ws);           // [B,H,S,64]   bf16, pre-scaled
  __bf16* kb  = (__bf16*)(ws + SZ);      // [B,H,S,64]   bf16
  __bf16* vtb = (__bf16*)(ws + 2 * SZ);  // [B,H,64,S]   bf16 (transposed)
  __bf16* ab  = (__bf16*)(ws + 3 * SZ);  // [B,S,1024]   bf16 (attn output)

  dim3 gp(M_TOT / 128, D_MODEL / 128);  // 32 x 8 blocks
  const float qscale = 0.125f;          // 1/sqrt(DEPTH), exact in bf16

  gemm_xwT<float, 0><<<gp, 256, 0, stream>>>(Q, Wq, bq, qb, qscale);
  gemm_xwT<float, 0><<<gp, 256, 0, stream>>>(K, Wk, bk, kb, 1.0f);
  gemm_xwT<float, 1><<<gp, 256, 0, stream>>>(V, Wv, bv, vtb, 1.0f);

  dim3 ga(SEQ / 64, BATCH * N_HEADS);  // 32 x 32 blocks
  attn_kernel<<<ga, 128, 0, stream>>>(qb, kb, vtb, ab);

  gemm_xwT<__bf16, 2><<<gp, 256, 0, stream>>>(ab, Wo, bo, out, 1.0f);
}
